// CasualAttention_37366215475931
// MI455X (gfx1250) — matrix operand order repaired and checked by reading, not compiled, not run on hardware
//
#include <hip/hip_runtime.h>

// Causal attention (QKV proj + softmax(QK^T)V), MI455X / gfx1250.
// Strategy: bf16 WMMA (v_wmma_f32_16x16x32_bf16) with f32 accumulation.
//   - 1/sqrt(D) folded into Q at projection time.
//   - V stored transposed so PV B-fragments are contiguous.
//   - Block-causal tile skipping (only j-tile <= q-tile computed).
//   - GEMM kernels block 4 N-tiles per wave: one A-fragment feeds 4 chained
//     WMMAs per K-step (4x less A traffic, denser matrix-op issue).
// Workspace layout (bytes):
//   [0,16M)     Q   bf16 [8192][1024]
//   [16M,32M)   K   bf16 [8192][1024]
//   [32M,48M)   Vt  bf16 [4][1024][2048]   (transposed V)
//   [48M,112M)  S   f32  [4][2048][2048]   (raw scaled scores)
//   [112M,144M) P   bf16 [4][2048][2048]   (exp(s - rowmax))
//   [144M,..)   linv f32 [8192]            (1 / rowsum)
// All intermediates (~144MB) fit in the 192MB L2, so HBM traffic is just
// x + W + out (~76MB).

#define D_IN  1024
#define D_OUT 1024
#define SEQ   2048
#define BATCH 4
#define NROWS (BATCH * SEQ) // 8192

typedef __attribute__((ext_vector_type(16))) __bf16 v16bf;
typedef __attribute__((ext_vector_type(8)))  float  v8f;

union BH8 { uint4 v; __bf16 h[8]; };

__device__ __forceinline__ __bf16 f2bf(float f) { return (__bf16)f; }

template <int O>
__device__ __forceinline__ void pack8f(v16bf& d, float4 lo, float4 hi) {
    d[O+0]=f2bf(lo.x); d[O+1]=f2bf(lo.y); d[O+2]=f2bf(lo.z); d[O+3]=f2bf(lo.w);
    d[O+4]=f2bf(hi.x); d[O+5]=f2bf(hi.y); d[O+6]=f2bf(hi.z); d[O+7]=f2bf(hi.w);
}

template <int O>
__device__ __forceinline__ void pack8h(v16bf& d, const BH8& u) {
#pragma unroll
    for (int i = 0; i < 8; ++i) d[O+i] = u.h[i];
}

__device__ __forceinline__ v8f wmma_bf16(v16bf a, v16bf b, v8f c) {
    return __builtin_amdgcn_wmma_f32_16x16x32_bf16(
        /*neg_a=*/false, a, /*neg_b=*/false, b,
        /*c_mod=*/(short)0, c, /*reuse_a=*/false, /*reuse_b=*/false);
}

// Build a B-fragment (32x16, K contiguous per lane) from a bf16 row pointer.
__device__ __forceinline__ v16bf load_bfragh(const __bf16* row, int off) {
    BH8 u0, u1;
    u0.v = *(const uint4*)(row + off);
    u1.v = *(const uint4*)(row + off + 8);
    v16bf B;
    pack8h<0>(B, u0); pack8h<8>(B, u1);
    return B;
}

// ---------------------------------------------------------------------------
// Kernel 1: QKV projections  y = x @ W^T, output bf16 (V transposed).
// One wave = one 16x64 output strip (4 N-tiles); K swept 32 at a time.
// ---------------------------------------------------------------------------
__global__ __launch_bounds__(128) void qkv_wmma_gemm(
    const float* __restrict__ x,
    const float* __restrict__ Wq,
    const float* __restrict__ Wk,
    const float* __restrict__ Wv,
    __bf16* __restrict__ Qb,
    __bf16* __restrict__ Kb,
    __bf16* __restrict__ Vt)
{
    const int lane  = threadIdx.x & 31;
    const int wave  = threadIdx.x >> 5;
    const int mtile = blockIdx.x;                    // 0..511 (8192 rows / 16)
    const int nbase = (blockIdx.y * 4 + wave) * 4;   // first of 4 N-tiles
    const int z     = blockIdx.z;                    // 0:Q 1:K 2:V

    const float* W = (z == 0) ? Wq : (z == 1) ? Wk : Wv;

    const int m  = lane & 15;   // A row within tile / B column within tile
    const int hg = lane >> 4;   // K-group for fragments, row-half for C

    const float* xrow = x + (size_t)(mtile * 16 + m) * D_IN;
    const float* wrow0 = W + (size_t)((nbase + 0) * 16 + m) * D_IN;
    const float* wrow1 = W + (size_t)((nbase + 1) * 16 + m) * D_IN;
    const float* wrow2 = W + (size_t)((nbase + 2) * 16 + m) * D_IN;
    const float* wrow3 = W + (size_t)((nbase + 3) * 16 + m) * D_IN;

    v8f acc0 = {}, acc1 = {}, acc2 = {}, acc3 = {};
    for (int kb = 0; kb < D_IN; kb += 32) {
        // A fragment (16x32 bf16 layout): K = 16*hg + j (contiguous)
        float4 a0 = *(const float4*)(xrow + kb + 16 * hg);
        float4 a1 = *(const float4*)(xrow + kb + 16 * hg + 4);
        float4 a2 = *(const float4*)(xrow + kb + 16 * hg + 8);
        float4 a3 = *(const float4*)(xrow + kb + 16 * hg + 12);
        v16bf A;
        pack8f<0>(A, a0, a1); pack8f<8>(A, a2, a3);

        // 4 B fragments (32x16): B[k][n] = W[n][k]; K = 16*hg + j (contiguous)
        const int wo = kb + 16 * hg;
        v16bf B0, B1, B2, B3;
        {
            float4 c0 = *(const float4*)(wrow0 + wo);
            float4 c1 = *(const float4*)(wrow0 + wo + 4);
            float4 c2 = *(const float4*)(wrow0 + wo + 8);
            float4 c3 = *(const float4*)(wrow0 + wo + 12);
            pack8f<0>(B0, c0, c1); pack8f<8>(B0, c2, c3);
        }
        {
            float4 c0 = *(const float4*)(wrow1 + wo);
            float4 c1 = *(const float4*)(wrow1 + wo + 4);
            float4 c2 = *(const float4*)(wrow1 + wo + 8);
            float4 c3 = *(const float4*)(wrow1 + wo + 12);
            pack8f<0>(B1, c0, c1); pack8f<8>(B1, c2, c3);
        }
        {
            float4 c0 = *(const float4*)(wrow2 + wo);
            float4 c1 = *(const float4*)(wrow2 + wo + 4);
            float4 c2 = *(const float4*)(wrow2 + wo + 8);
            float4 c3 = *(const float4*)(wrow2 + wo + 12);
            pack8f<0>(B2, c0, c1); pack8f<8>(B2, c2, c3);
        }
        {
            float4 c0 = *(const float4*)(wrow3 + wo);
            float4 c1 = *(const float4*)(wrow3 + wo + 4);
            float4 c2 = *(const float4*)(wrow3 + wo + 8);
            float4 c3 = *(const float4*)(wrow3 + wo + 12);
            pack8f<0>(B3, c0, c1); pack8f<8>(B3, c2, c3);
        }

        acc0 = wmma_bf16(A, B0, acc0);
        acc1 = wmma_bf16(A, B1, acc1);
        acc2 = wmma_bf16(A, B2, acc2);
        acc3 = wmma_bf16(A, B3, acc3);
    }

    // C layout: row = r + 8*hg, col = m. Fold 1/sqrt(D)=1/32 into Q.
    const float scale = (z == 0) ? 0.03125f : 1.0f;
#pragma unroll
    for (int t = 0; t < 4; ++t) {
        const v8f& acc = (t == 0) ? acc0 : (t == 1) ? acc1 : (t == 2) ? acc2 : acc3;
        const int ncol = (nbase + t) * 16 + m;
#pragma unroll
        for (int r = 0; r < 8; ++r) {
            const int row = mtile * 16 + r + 8 * hg;   // global row in [0,8192)
            const float v = acc[r] * scale;
            if (z == 0) {
                Qb[(size_t)row * D_OUT + ncol] = f2bf(v);
            } else if (z == 1) {
                Kb[(size_t)row * D_OUT + ncol] = f2bf(v);
            } else {
                const int b = row >> 11;       // row / SEQ
                const int s = row & (SEQ - 1);
                Vt[((size_t)b * D_OUT + ncol) * SEQ + s] = f2bf(v);
            }
        }
    }
}

// ---------------------------------------------------------------------------
// Kernel 2: block-causal scores (WMMA) + softmax (max/exp/sum) -> bf16 P.
// One wave = one 16-row query tile; sweeps key tiles 0..qt.
// ---------------------------------------------------------------------------
__global__ __launch_bounds__(128) void scores_softmax(
    const __bf16* __restrict__ Qb,
    const __bf16* __restrict__ Kb,
    float*  __restrict__ Sc,
    __bf16* __restrict__ P,
    float*  __restrict__ linv)
{
    const int lane = threadIdx.x & 31;
    const int wave = threadIdx.x >> 5;
    const int wid  = blockIdx.x * 4 + wave;  // 0..511
    const int b    = wid >> 7;               // batch
    const int qt   = wid & 127;              // query tile 0..127

    const int m  = lane & 15;
    const int hg = lane >> 4;

    const __bf16* qrow = Qb + (size_t)(b * SEQ + qt * 16 + m) * D_OUT;
    float*  scb = Sc + (size_t)b * SEQ * SEQ;
    __bf16* pb  = P  + (size_t)b * SEQ * SEQ;

    float mrow[8];
#pragma unroll
    for (int r = 0; r < 8; ++r) mrow[r] = -__builtin_inff();

    for (int jt = 0; jt <= qt; ++jt) {
        const __bf16* krow = Kb + (size_t)(b * SEQ + jt * 16 + m) * D_OUT;
        v8f acc = {};
        for (int eb = 0; eb < D_OUT; eb += 32) {
            BH8 qa0, qa1;
            qa0.v = *(const uint4*)(qrow + eb + 16 * hg);
            qa1.v = *(const uint4*)(qrow + eb + 16 * hg + 8);
            v16bf A;
            pack8h<0>(A, qa0); pack8h<8>(A, qa1);
            v16bf B = load_bfragh(krow, eb + 16 * hg);
            acc = wmma_bf16(A, B, acc);
        }
        if (jt == qt) {            // causal mask on diagonal tile
#pragma unroll
            for (int r = 0; r < 8; ++r) {
                const int qi = r + 8 * hg;
                acc[r] = (m > qi) ? -__builtin_inff() : acc[r];
            }
        }
#pragma unroll
        for (int r = 0; r < 8; ++r) {
            const int qi = qt * 16 + r + 8 * hg;
            const int kj = jt * 16 + m;
            scb[(size_t)qi * SEQ + kj] = acc[r];
            mrow[r] = fmaxf(mrow[r], acc[r]);
        }
    }

    // Row max across the 16 lanes holding each C row (halves are independent).
#pragma unroll
    for (int r = 0; r < 8; ++r)
        for (int off = 1; off < 16; off <<= 1)
            mrow[r] = fmaxf(mrow[r], __shfl_xor(mrow[r], off, 32));

    float lsum[8];
#pragma unroll
    for (int r = 0; r < 8; ++r) lsum[r] = 0.0f;

    for (int jt = 0; jt <= qt; ++jt) {
#pragma unroll
        for (int r = 0; r < 8; ++r) {
            const int qi = qt * 16 + r + 8 * hg;
            const int kj = jt * 16 + m;
            const float s = scb[(size_t)qi * SEQ + kj];
            const float p = __expf(s - mrow[r]);   // exp(-inf - m) = 0
            lsum[r] += p;
            pb[(size_t)qi * SEQ + kj] = f2bf(p);
        }
    }
#pragma unroll
    for (int r = 0; r < 8; ++r)
        for (int off = 1; off < 16; off <<= 1)
            lsum[r] += __shfl_xor(lsum[r], off, 32);

    if (m == 0) {
#pragma unroll
        for (int r = 0; r < 8; ++r)
            linv[b * SEQ + qt * 16 + r + 8 * hg] = 1.0f / lsum[r];
    }

    // Zero-pad tile jt = qt+1 so the PV K-loop can stay 32-wide.
    if ((qt & 1) == 0) {
#pragma unroll
        for (int r = 0; r < 8; ++r) {
            const int qi = qt * 16 + r + 8 * hg;
            const int kj = (qt + 1) * 16 + m;
            pb[(size_t)qi * SEQ + kj] = f2bf(0.0f);
        }
    }
}

// ---------------------------------------------------------------------------
// Kernel 3: O = (P . V) * linv, WMMA over the causal K extent only.
// One wave = one 16x64 output strip (4 e-tiles) sharing the P A-fragment.
// ---------------------------------------------------------------------------
__global__ __launch_bounds__(128) void pv_wmma_gemm(
    const __bf16* __restrict__ P,
    const __bf16* __restrict__ Vt,
    const float*  __restrict__ linv,
    float* __restrict__ out)
{
    const int lane = threadIdx.x & 31;
    const int wave = threadIdx.x >> 5;
    const int qt    = blockIdx.x;                    // 0..127
    const int ebase = (blockIdx.y * 4 + wave) * 4;   // first of 4 e-tiles
    const int b     = blockIdx.z;

    const int m  = lane & 15;
    const int hg = lane >> 4;

    const __bf16* prow = P + (size_t)(b * SEQ + qt * 16 + m) * SEQ;
    const __bf16* vrow0 = Vt + (size_t)(b * D_OUT + (ebase + 0) * 16 + m) * SEQ;
    const __bf16* vrow1 = Vt + (size_t)(b * D_OUT + (ebase + 1) * 16 + m) * SEQ;
    const __bf16* vrow2 = Vt + (size_t)(b * D_OUT + (ebase + 2) * 16 + m) * SEQ;
    const __bf16* vrow3 = Vt + (size_t)(b * D_OUT + (ebase + 3) * 16 + m) * SEQ;

    const int kend = ((qt + 1) * 16 + 31) & ~31;   // causal extent, 32-aligned
    v8f acc0 = {}, acc1 = {}, acc2 = {}, acc3 = {};
    for (int kb = 0; kb < kend; kb += 32) {
        BH8 p0, p1;
        p0.v = *(const uint4*)(prow + kb + 16 * hg);
        p1.v = *(const uint4*)(prow + kb + 16 * hg + 8);
        v16bf A;
        pack8h<0>(A, p0); pack8h<8>(A, p1);

        const int vo = kb + 16 * hg;
        v16bf B0 = load_bfragh(vrow0, vo);
        v16bf B1 = load_bfragh(vrow1, vo);
        v16bf B2 = load_bfragh(vrow2, vo);
        v16bf B3 = load_bfragh(vrow3, vo);

        acc0 = wmma_bf16(A, B0, acc0);
        acc1 = wmma_bf16(A, B1, acc1);
        acc2 = wmma_bf16(A, B2, acc2);
        acc3 = wmma_bf16(A, B3, acc3);
    }

#pragma unroll
    for (int t = 0; t < 4; ++t) {
        const v8f& acc = (t == 0) ? acc0 : (t == 1) ? acc1 : (t == 2) ? acc2 : acc3;
        const int e = (ebase + t) * 16 + m;
#pragma unroll
        for (int r = 0; r < 8; ++r) {
            const int qi = qt * 16 + r + 8 * hg;
            out[(size_t)(b * SEQ + qi) * D_OUT + e] = acc[r] * linv[b * SEQ + qi];
        }
    }
}

// ---------------------------------------------------------------------------
extern "C" void kernel_launch(void* const* d_in, const int* in_sizes, int n_in,
                              void* d_out, int out_size, void* d_ws, size_t ws_size,
                              hipStream_t stream) {
    (void)in_sizes; (void)n_in; (void)out_size; (void)ws_size;

    const float* x  = (const float*)d_in[0];
    const float* Wq = (const float*)d_in[1];
    const float* Wk = (const float*)d_in[2];
    const float* Wv = (const float*)d_in[3];
    float* out = (float*)d_out;

    char* ws = (char*)d_ws;
    const size_t MB = 1024u * 1024u;
    __bf16* Qb   = (__bf16*)(ws);
    __bf16* Kb   = (__bf16*)(ws + 16 * MB);
    __bf16* Vt   = (__bf16*)(ws + 32 * MB);
    float*  Sc   = (float*) (ws + 48 * MB);
    __bf16* Pp   = (__bf16*)(ws + 112 * MB);
    float*  linv = (float*) (ws + 144 * MB);

    const dim3 blk(128, 1, 1);
    // 512 row-tiles x (16 N-groups of 4 tiles -> 4 blocks of 4 waves) x {Q,K,V}
    qkv_wmma_gemm<<<dim3(NROWS / 16, 4, 3), blk, 0, stream>>>(
        x, Wq, Wk, Wv, Qb, Kb, Vt);
    // 512 query tiles total (4 per block)
    scores_softmax<<<dim3((BATCH * (SEQ / 16)) / 4, 1, 1), blk, 0, stream>>>(
        Qb, Kb, Sc, Pp, linv);
    // 128 q-tiles x (16 e-groups of 4 tiles -> 4 blocks of 4 waves) x 4 batches
    pv_wmma_gemm<<<dim3(SEQ / 16, 4, BATCH), blk, 0, stream>>>(
        Pp, Vt, linv, out);
}
